// ComplexDepthwiseBatchNorm_26182120636725
// MI455X (gfx1250) — matrix operand order repaired and checked by reading, not compiled, not run on hardware
//
#include <hip/hip_runtime.h>

typedef __attribute__((ext_vector_type(16))) _Float16 v16h;
typedef __attribute__((ext_vector_type(8)))  float    v8f;
typedef __attribute__((ext_vector_type(4)))  float    v4f;

constexpr int   kN            = 16384;
constexpr int   kC            = 8;
constexpr int   kF            = 257;
constexpr int   kCols         = kC * kF;            // 2056
constexpr int   kTiles        = (kCols + 15) / 16;  // 129
constexpr int   kColsPad      = kTiles * 16;        // 2064
constexpr int   kChunks       = 32;                 // row chunks
constexpr int   kRowsPerChunk = kN / kChunks;       // 512
constexpr int   kWavesPerBlk  = 8;
constexpr float kEps          = 1e-6f;
constexpr float kDeltaMax     = 1e8f;

// ---------------------------------------------------------------------------
// Pass 1: per-column 5-moment partial sums via WMMA ones-trick.
// One wave handles a 16-column tile over a 512-row chunk, 32 rows per step.
// B-matrix layout (32x16 f16): lane l = column n=l%16, element s = row K=16*(l/16)+s.
// A = ones(16x32)  =>  D[0][n] = sum_k B[k][n]  (row-0 of C accumulates).
// ---------------------------------------------------------------------------
__global__ __launch_bounds__(256) void cbn_stats_wmma(
    const float* __restrict__ xr, const float* __restrict__ xi,
    float* __restrict__ P) {
  const int tile  = blockIdx.x;                       // 0..128
  const int wave  = threadIdx.x >> 5;
  const int lane  = threadIdx.x & 31;
  const int chunk = blockIdx.y * kWavesPerBlk + wave; // 0..31
  const int col0  = tile * 16;
  const bool full = (col0 + 16) <= kCols;

  v16h ones;
#pragma unroll
  for (int h = 0; h < 16; ++h) ones[h] = (_Float16)1.0f;

  v8f c_r = {}, c_i = {}, c_rr = {}, c_ii = {}, c_ri = {};

  const int row0 = chunk * kRowsPerChunk;
#pragma unroll 4
  for (int it = 0; it < kRowsPerChunk / 32; ++it) {
    const int row = row0 + it * 32 + 16 * (lane >> 4); // element s == K row 16h+s of B
    const float* pr = xr + (size_t)row * kCols + col0 + (lane & 15);
    const float* pi = xi + (size_t)row * kCols + col0 + (lane & 15);

    float fr[16], fi[16];
    {
      const bool ok = full || (col0 + (lane & 15)) < kCols;
#pragma unroll
      for (int h = 0; h < 16; ++h) {
        fr[h] = ok ? pr[(size_t)h * kCols] : 0.0f;
        fi[h] = ok ? pi[(size_t)h * kCols] : 0.0f;
      }
    }

    v16h hr, hi, hrr, hii, hri;
#pragma unroll
    for (int h = 0; h < 16; ++h) {
      const float a = fr[h], b = fi[h];
      hr[h]  = (_Float16)a;
      hi[h]  = (_Float16)b;
      hrr[h] = (_Float16)(a * a);
      hii[h] = (_Float16)(b * b);
      hri[h] = (_Float16)(a * b);
    }
    // Five independent accumulation chains -> latency overlap in the XDL pipe.
    c_r  = __builtin_amdgcn_wmma_f32_16x16x32_f16(false, ones, false, hr,  (short)0, c_r,  false, false);
    c_i  = __builtin_amdgcn_wmma_f32_16x16x32_f16(false, ones, false, hi,  (short)0, c_i,  false, false);
    c_rr = __builtin_amdgcn_wmma_f32_16x16x32_f16(false, ones, false, hrr, (short)0, c_rr, false, false);
    c_ii = __builtin_amdgcn_wmma_f32_16x16x32_f16(false, ones, false, hii, (short)0, c_ii, false, false);
    c_ri = __builtin_amdgcn_wmma_f32_16x16x32_f16(false, ones, false, hri, (short)0, c_ri, false, false);
  }

  // D row 0 lives in c[0], lanes 0..15 (lane == column n).
  if (lane < 16) {
    const size_t col = (size_t)tile * 16 + lane;
    P[((size_t)0 * kChunks + chunk) * kColsPad + col] = c_r[0];
    P[((size_t)1 * kChunks + chunk) * kColsPad + col] = c_i[0];
    P[((size_t)2 * kChunks + chunk) * kColsPad + col] = c_rr[0];
    P[((size_t)3 * kChunks + chunk) * kColsPad + col] = c_ii[0];
    P[((size_t)4 * kChunks + chunk) * kColsPad + col] = c_ri[0];
  }
}

// ---------------------------------------------------------------------------
// Pass 2: deterministic fold of partials + whitening + affine fold.
// Emits per-column constants: [Zrr, Zri, Zir, Zii, Cr, Ci] (plane layout).
// ---------------------------------------------------------------------------
__global__ __launch_bounds__(256) void cbn_finalize(
    const float* __restrict__ P,
    const float* __restrict__ Wrr, const float* __restrict__ Wri,
    const float* __restrict__ Wii, const float* __restrict__ Br,
    const float* __restrict__ Bi, float* __restrict__ T) {
  const int j = blockIdx.x * blockDim.x + threadIdx.x;
  if (j >= kCols) return;

  float s0 = 0.f, s1 = 0.f, s2 = 0.f, s3 = 0.f, s4 = 0.f;
  for (int ch = 0; ch < kChunks; ++ch) {
    s0 += P[((size_t)0 * kChunks + ch) * kColsPad + j];
    s1 += P[((size_t)1 * kChunks + ch) * kColsPad + j];
    s2 += P[((size_t)2 * kChunks + ch) * kColsPad + j];
    s3 += P[((size_t)3 * kChunks + ch) * kColsPad + j];
    s4 += P[((size_t)4 * kChunks + ch) * kColsPad + j];
  }
  const float invN = 1.0f / (float)kN;
  const float Mr  = s0 * invN;
  const float Mi  = s1 * invN;
  const float Vrr = s2 * invN - Mr * Mr;
  const float Vii = s3 * invN - Mi * Mi;
  const float Vri = s4 * invN - Mr * Mi;

  const float tau   = Vrr + Vii;
  const float delta = fminf(fmaxf(Vrr * Vii - Vri * Vri, kEps), kDeltaMax);
  const float s     = sqrtf(delta);
  const float t     = sqrtf(tau + 2.0f * s);
  const float rst   = 1.0f / (s * t);
  const float Urr   = (s + Vii) * rst;
  const float Uii   = (s + Vrr) * rst;
  const float Uri   = -Vri * rst;

  const float wrr = Wrr[j], wri = Wri[j], wii = Wii[j];
  const float Zrr = wrr * Urr + wri * Uri;
  const float Zri = wrr * Uri + wri * Uii;
  const float Zir = wri * Urr + wii * Uri;
  const float Zii = wri * Uri + wii * Uii;
  // Fold mean subtraction and bias into additive constants.
  const float Cr = Br[j] - Zrr * Mr - Zri * Mi;
  const float Ci = Bi[j] - Zir * Mr - Zii * Mi;

  T[0 * kCols + j] = Zrr;
  T[1 * kCols + j] = Zri;
  T[2 * kCols + j] = Zir;
  T[3 * kCols + j] = Zii;
  T[4 * kCols + j] = Cr;
  T[5 * kCols + j] = Ci;
}

// ---------------------------------------------------------------------------
// Pass 3: streaming y = Z*x + c, float4 (b128) in/out. 2056 % 4 == 0 so a
// float4 never crosses a row; coefficient table (48 KB) is cache resident.
// ---------------------------------------------------------------------------
__global__ __launch_bounds__(256) void cbn_apply(
    const float* __restrict__ xr, const float* __restrict__ xi,
    const float* __restrict__ T, float* __restrict__ yr,
    float* __restrict__ yi) {
  const size_t e0  = ((size_t)blockIdx.x * blockDim.x + threadIdx.x) * 4;
  const int    col = (int)(e0 % (size_t)kCols);

  const v4f r = *(const v4f*)(xr + e0);
  const v4f i = *(const v4f*)(xi + e0);
  v4f outr, outi;
#pragma unroll
  for (int j = 0; j < 4; ++j) {
    const int c = col + j;
    const float zrr = T[0 * kCols + c];
    const float zri = T[1 * kCols + c];
    const float zir = T[2 * kCols + c];
    const float zii = T[3 * kCols + c];
    const float cr  = T[4 * kCols + c];
    const float ci  = T[5 * kCols + c];
    outr[j] = fmaf(zrr, r[j], fmaf(zri, i[j], cr));
    outi[j] = fmaf(zir, r[j], fmaf(zii, i[j], ci));
  }
  *(v4f*)(yr + e0) = outr;
  *(v4f*)(yi + e0) = outi;
}

// ---------------------------------------------------------------------------
extern "C" void kernel_launch(void* const* d_in, const int* in_sizes, int n_in,
                              void* d_out, int out_size, void* d_ws, size_t ws_size,
                              hipStream_t stream) {
  (void)in_sizes; (void)n_in; (void)out_size; (void)ws_size;
  const float* xr  = (const float*)d_in[0];
  const float* xi  = (const float*)d_in[1];
  const float* Wrr = (const float*)d_in[2];
  const float* Wri = (const float*)d_in[3];
  const float* Wii = (const float*)d_in[4];
  const float* Br  = (const float*)d_in[5];
  const float* Bi  = (const float*)d_in[6];

  float* out = (float*)d_out;
  float* P   = (float*)d_ws;                          // 5*32*2064 f32 ≈ 1.3 MB
  float* T   = P + (size_t)5 * kChunks * kColsPad;    // 6*2056 f32

  // Pass 1: 129 col-tiles x (32 chunks / 8 waves-per-block)
  dim3 g1(kTiles, kChunks / kWavesPerBlk);
  cbn_stats_wmma<<<g1, 256, 0, stream>>>(xr, xi, P);

  // Pass 2: one thread per column
  cbn_finalize<<<(kCols + 255) / 256, 256, 0, stream>>>(P, Wrr, Wri, Wii, Br, Bi, T);

  // Pass 3: N*kCols elements / 4 per thread / 256 per block = 32896 blocks exact
  const size_t total = (size_t)kN * kCols;            // 33,685,504
  cbn_apply<<<(unsigned)(total / 4 / 256), 256, 0, stream>>>(
      xr, xi, T, out, out + total);
}
